// Second_Order_PAM_Module_77653008711848
// MI455X (gfx1250) — matrix operand order repaired and checked by reading, not compiled, not run on hardware
//
#include <hip/hip_runtime.h>

typedef __attribute__((ext_vector_type(16))) __bf16 bf16x16;
typedef __attribute__((ext_vector_type(8)))  float  v8f;
typedef __attribute__((ext_vector_type(4)))  unsigned int u32x4;
typedef __attribute__((ext_vector_type(8)))  int i32x8;
typedef __attribute__((ext_vector_type(4)))  int i32x4;

#define B_  8
#define C_  512
#define CQ_ 64
#define N_  2304   // 48*48

// ---------- helpers ----------

__device__ inline unsigned short f2bf(float f) {
    union { float f; unsigned u; } x; x.f = f;
    unsigned r = x.u + 0x7FFFu + ((x.u >> 16) & 1u);   // round-to-nearest-even
    return (unsigned short)(r >> 16);
}

union FragU {
    bf16x16 v;
    uint4   q[2];
};

// A fragment, 16x32 bf16, source row-major [row][K], stride ld (elements).
// Lane r (<16): K 0..15 of row r ; lane r+16: K 16..31 of row r.
__device__ inline bf16x16 load_frag_A(const unsigned short* base, int ld) {
    const int lane = threadIdx.x & 31;
    const int r  = lane & 15;
    const int hf = lane >> 4;
    const unsigned short* p = base + (size_t)r * ld + hf * 16;
    FragU f;
    f.q[0] = *(const uint4*)(p);
    f.q[1] = *(const uint4*)(p + 8);
    return f.v;
}

// B fragment, 32x16 bf16, source stored as [col][K], stride ld.
// Lane r (<16): K 0..15 of col r ; lane r+16: K 16..31 of col r.
__device__ inline bf16x16 load_frag_B(const unsigned short* base, int ld) {
    const int lane = threadIdx.x & 31;
    const int r  = lane & 15;
    const int hf = lane >> 4;
    const unsigned short* p = base + (size_t)r * ld + hf * 16;
    FragU f;
    f.q[0] = *(const uint4*)(p);
    f.q[1] = *(const uint4*)(p + 8);
    return f.v;
}

__device__ inline v8f wmma_bf16(bf16x16 a, bf16x16 b, v8f c) {
    return __builtin_amdgcn_wmma_f32_16x16x32_bf16(false, a, false, b, (short)0, c,
                                                   false, false);
}

// ---- Tensor Data Mover: 2D tile Global -> LDS (D# per ISA ch.8) ----
// data_size = 2 bytes. Tensor [tensor_d1][tensor_d0], row stride stride0 elements.
// Tile [tile_d1][tile_d0] starting at gaddr, packed row-major into LDS at lds_off.
__device__ inline void tdm_load_2d(unsigned lds_off, const void* gaddr,
                                   unsigned tile_d0, unsigned tile_d1,
                                   unsigned tensor_d0, unsigned tensor_d1,
                                   unsigned long long stride0) {
#if __has_builtin(__builtin_amdgcn_tensor_load_to_lds)
    unsigned long long ga = (unsigned long long)gaddr;
    u32x4 g0;
    g0[0] = 1u;                                            // count=1, user D#
    g0[1] = lds_off;                                       // lds_addr  (bits 63:32)
    g0[2] = (unsigned)(ga & 0xFFFFFFFFull);                // global_addr[31:0]
    g0[3] = (unsigned)((ga >> 32) & 0x1FFFFFFull)          // global_addr[56:32]
            | (2u << 30);                                  // type = 2 (image)
    i32x8 g1;
    g1[0] = (int)(1u << 16);                               // data_size=1 -> 2B; mask=0
    g1[1] = (int)((tensor_d0 & 0xFFFFu) << 16);            // tensor_dim0[15:0]
    g1[2] = (int)(((tensor_d0 >> 16) & 0xFFFFu)            // tensor_dim0[31:16]
            | ((tensor_d1 & 0xFFFFu) << 16));              // tensor_dim1[15:0]
    g1[3] = (int)(((tensor_d1 >> 16) & 0xFFFFu)            // tensor_dim1[31:16]
            | ((tile_d0 & 0xFFFFu) << 16));                // tile_dim0
    g1[4] = (int)(tile_d1 & 0xFFFFu);                      // tile_dim1 (tile_dim2=0)
    g1[5] = (int)(stride0 & 0xFFFFFFFFull);                // tensor_dim0_stride[31:0]
    g1[6] = (int)((stride0 >> 32) & 0xFFFFull);            // stride[47:32]
    g1[7] = 0;
    i32x4 z4 = {0, 0, 0, 0};                               // D# group 2 (unused, 2D)
    i32x8 z8 = {0, 0, 0, 0, 0, 0, 0, 0};                   // D# group 3 (unused, 2D)
    __builtin_amdgcn_tensor_load_to_lds(g0, g1, z4, z4, z8, 0);
#else
    (void)lds_off; (void)gaddr; (void)tile_d0; (void)tile_d1;
    (void)tensor_d0; (void)tensor_d1; (void)stride0;
#endif
}

template <int N>
__device__ inline void wait_tensorcnt() {
#if __has_builtin(__builtin_amdgcn_s_wait_tensorcnt)
    __builtin_amdgcn_s_wait_tensorcnt(N);   // immediate: must be an ICE
#endif
}

__device__ inline unsigned lds_offset_of(const void* p) {
    // generic pointer to LDS: low 32 bits carry the LDS byte offset
    return (unsigned)(unsigned long long)p;
}

// ---------- stage 0: precision conversion / transpose ----------

__global__ __launch_bounds__(256) void cvt_f32_bf16(const float* __restrict__ in,
                                                    unsigned short* __restrict__ out,
                                                    int n) {
    int i = blockIdx.x * 256 + threadIdx.x;
    if (i < n) out[i] = f2bf(in[i]);
}

// xT[b][n][c] = x[b][c][n], bf16
__global__ __launch_bounds__(256) void transpose_x(const float* __restrict__ x,
                                                   unsigned short* __restrict__ xT) {
    size_t i = (size_t)blockIdx.x * 256 + threadIdx.x;   // = b*N*C + n*C + c
    int c = (int)(i % C_);
    size_t t = i / C_;
    int n = (int)(t % N_);
    int b = (int)(t / N_);
    xT[i] = f2bf(x[((size_t)b * C_ + c) * N_ + n]);
}

// ---------- stage 1: q/k projection, fused center+scale ----------
// One wave per (b, 16-row n-tile). D tile = [n][o]; Cq = 64 -> 4 o-tiles.
__global__ __launch_bounds__(32) void qk_proj(const unsigned short* __restrict__ xT,
                                              const unsigned short* __restrict__ Wqb,
                                              const unsigned short* __restrict__ Wkb,
                                              const float* __restrict__ bq,
                                              const float* __restrict__ bk,
                                              unsigned short* __restrict__ qcT,
                                              unsigned short* __restrict__ kT) {
    const int blk = blockIdx.x;
    const int b  = blk / (N_ / 16);
    const int n0 = (blk % (N_ / 16)) * 16;
    const int lane = threadIdx.x & 31;
    const int r  = lane & 15;
    const int hf = lane >> 4;

    const unsigned short* xrow = xT + ((size_t)b * N_ + n0) * C_;

    v8f qa[4] = {};
    v8f ka[4] = {};
#pragma unroll 2
    for (int c0 = 0; c0 < C_; c0 += 32) {
        bf16x16 a = load_frag_A(xrow + c0, C_);
        bf16x16 bq4[4], bk4[4];
#pragma unroll
        for (int t = 0; t < 4; ++t) {
            bq4[t] = load_frag_B(Wqb + (size_t)(t * 16) * C_ + c0, C_);
            bk4[t] = load_frag_B(Wkb + (size_t)(t * 16) * C_ + c0, C_);
        }
#pragma unroll
        for (int t = 0; t < 4; ++t) {
            qa[t] = wmma_bf16(a, bq4[t], qa[t]);
            ka[t] = wmma_bf16(a, bk4[t], ka[t]);
        }
    }
    // bias (per output channel o = t*16 + r, uniform over rows)
#pragma unroll
    for (int t = 0; t < 4; ++t) {
        float bqv = bq[t * 16 + r];
        float bkv = bk[t * 16 + r];
#pragma unroll
        for (int i = 0; i < 8; ++i) { qa[t][i] += bqv; ka[t][i] += bkv; }
    }
    // center q over Cq, apply 1/(Cq*sqrt(Cq)) = 1/512, emit bf16
    const float scl = 1.0f / 512.0f;
#pragma unroll
    for (int i = 0; i < 8; ++i) {
        float s = qa[0][i] + qa[1][i] + qa[2][i] + qa[3][i];
        s += __shfl_xor(s, 1, 32);
        s += __shfl_xor(s, 2, 32);
        s += __shfl_xor(s, 4, 32);
        s += __shfl_xor(s, 8, 32);
        float mean = s * (1.0f / 64.0f);
        int n = n0 + i + 8 * hf;
        size_t rowoff = ((size_t)b * N_ + n) * CQ_;
#pragma unroll
        for (int t = 0; t < 4; ++t) {
            int o = t * 16 + r;
            qcT[rowoff + o] = f2bf((qa[t][i] - mean) * scl);
            kT [rowoff + o] = f2bf(ka[t][i]);
        }
    }
}

// ---------- stage 2: v projection (software-pipelined) ----------
// 4 waves per block; block covers 64 o x 64 n. D tile = [o][n].
__global__ __launch_bounds__(128) void v_proj(const unsigned short* __restrict__ xT,
                                              const unsigned short* __restrict__ Wvb,
                                              const float* __restrict__ bv,
                                              unsigned short* __restrict__ vbf) {
    const int blk = blockIdx.x;
    const int nb  = blk % (N_ / 64);
    const int t2  = blk / (N_ / 64);
    const int ob  = t2 % (C_ / 64);
    const int b   = t2 / (C_ / 64);
    const int w    = threadIdx.x >> 5;
    const int lane = threadIdx.x & 31;
    const int r  = lane & 15;
    const int hf = lane >> 4;
    const int o0 = ob * 64 + w * 16;
    const int n0 = nb * 64;

    const unsigned short* wrow  = Wvb + (size_t)o0 * C_;
    const unsigned short* xbase = xT + (size_t)b * N_ * C_;

    v8f acc[4] = {};
    // prologue: fragments for K-step 0
    bf16x16 a_cur = load_frag_A(wrow, C_);
    bf16x16 bx_cur[4];
#pragma unroll
    for (int t = 0; t < 4; ++t)
        bx_cur[t] = load_frag_B(xbase + (size_t)(n0 + t * 16) * C_, C_);

    for (int c0 = 0; c0 < C_; c0 += 32) {
        // start next K-step's loads before consuming the current fragments
        const int c1 = (c0 + 32 < C_) ? (c0 + 32) : c0;
        bf16x16 a_nxt = load_frag_A(wrow + c1, C_);
        bf16x16 bx_nxt[4];
#pragma unroll
        for (int t = 0; t < 4; ++t)
            bx_nxt[t] = load_frag_B(xbase + (size_t)(n0 + t * 16) * C_ + c1, C_);
#pragma unroll
        for (int t = 0; t < 4; ++t)
            acc[t] = wmma_bf16(a_cur, bx_cur[t], acc[t]);
        a_cur = a_nxt;
#pragma unroll
        for (int t = 0; t < 4; ++t) bx_cur[t] = bx_nxt[t];
    }
#pragma unroll
    for (int i = 0; i < 8; ++i) {
        int o = o0 + i + 8 * hf;
        float bias = bv[o];
        size_t rowoff = ((size_t)b * C_ + o) * N_;
#pragma unroll
        for (int t = 0; t < 4; ++t) {
            int n = n0 + t * 16 + r;
            vbf[rowoff + n] = f2bf(acc[t][i] + bias);
        }
    }
}

// ---------- stage 3: streaming (flash) attention + residual ----------
// Block = 4 waves per (b, 16-row n-tile); wave w owns channels [w*128, w*128+128).
// K/V panels double-buffered in LDS; the Tensor Data Mover streams block i+1
// while the WMMAs consume block i (s_wait_tensorcnt 2 = previous pair landed).
__global__ __launch_bounds__(128) void attn(const unsigned short* __restrict__ qcT,
                                            const unsigned short* __restrict__ kT,
                                            const unsigned short* __restrict__ vbf,
                                            const float* __restrict__ x,
                                            const float* __restrict__ gamma,
                                            float* __restrict__ out) {
    __shared__ __align__(16) unsigned short Kbuf[2][32][CQ_];   // [m_local][o]   8 KB
    __shared__ __align__(16) unsigned short Vbuf[2][C_][32];    // [c][m_local]  64 KB
    __shared__ __align__(16) unsigned short Plds[4][16][32];    // per-wave P     4 KB

    const int blk = blockIdx.x;
    const int b  = blk / (N_ / 16);
    const int n0 = (blk % (N_ / 16)) * 16;
    const int w    = threadIdx.x >> 5;
    const int lane = threadIdx.x & 31;
    const int r  = lane & 15;
    const int hf = lane >> 4;
    const int c0 = w * 128;

    const unsigned short* qrow  = qcT + ((size_t)b * N_ + n0) * CQ_;
    const unsigned short* kbase = kT  + (size_t)b * N_ * CQ_;
    const unsigned short* vbase = vbf + (size_t)b * C_ * N_;

    bf16x16 aq0 = load_frag_A(qrow,      CQ_);   // K = 0..31 of qc
    bf16x16 aq1 = load_frag_A(qrow + 32, CQ_);   // K = 32..63

    const unsigned kOff0 = lds_offset_of(&Kbuf[0][0][0]);
    const unsigned kOff1 = lds_offset_of(&Kbuf[1][0][0]);
    const unsigned vOff0 = lds_offset_of(&Vbuf[0][0][0]);
    const unsigned vOff1 = lds_offset_of(&Vbuf[1][0][0]);

    v8f O[8] = {};
    float rm[8], rl[8];
#pragma unroll
    for (int i = 0; i < 8; ++i) { rm[i] = -1e30f; rl[i] = 0.0f; }

    const int nblk = N_ / 32;
#if __has_builtin(__builtin_amdgcn_tensor_load_to_lds)
    if (w == 0) {   // prologue: stream block 0 into buffer 0
        tdm_load_2d(kOff0, kbase, CQ_, 32, CQ_, N_, CQ_);
        tdm_load_2d(vOff0, vbase, 32, C_, N_, C_, N_);
    }
#endif

    for (int ib = 0; ib < nblk; ++ib) {
        const int m0  = ib * 32;
        const int cur = ib & 1;
        __syncthreads();   // all reads of the other buffer (iter ib-1) retired
#if __has_builtin(__builtin_amdgcn_tensor_load_to_lds)
        if (w == 0) {
            if (ib + 1 < nblk) {
                const unsigned kN = cur ? kOff0 : kOff1;
                const unsigned vN = cur ? vOff0 : vOff1;
                const int m1 = m0 + 32;
                tdm_load_2d(kN, kbase + (size_t)m1 * CQ_, CQ_, 32, CQ_, N_, CQ_);
                tdm_load_2d(vN, vbase + m1, 32, C_, N_, C_, N_);
                wait_tensorcnt<2>();   // in-order retire: current pair has landed
            } else {
                wait_tensorcnt<0>();
            }
        }
#else
        {   // cooperative fallback copy of the current block (no async engine)
            const int tid = threadIdx.x;
            const uint4* ks = (const uint4*)(kbase + (size_t)m0 * CQ_);
            uint4* kd = (uint4*)&Kbuf[cur][0][0];
            for (int i = tid; i < (32 * CQ_) / 8; i += 128) kd[i] = ks[i];
            for (int i = tid; i < C_ * 4; i += 128) {
                int c = i >> 2, j = i & 3;
                ((uint4*)&Vbuf[cur][c][0])[j] =
                    *(const uint4*)(vbase + (size_t)c * N_ + m0 + j * 8);
            }
        }
#endif
        __syncthreads();   // current panels visible to all 4 waves

        // S = qc @ k : two 16x16 tiles over 32 m-columns, K = 64 in two steps
        FragU kb[4];
#pragma unroll
        for (int t = 0; t < 2; ++t) {
            kb[2 * t].q[0]     = *(const uint4*)(&Kbuf[cur][t * 16 + r][hf * 16]);
            kb[2 * t].q[1]     = *(const uint4*)(&Kbuf[cur][t * 16 + r][hf * 16 + 8]);
            kb[2 * t + 1].q[0] = *(const uint4*)(&Kbuf[cur][t * 16 + r][32 + hf * 16]);
            kb[2 * t + 1].q[1] = *(const uint4*)(&Kbuf[cur][t * 16 + r][32 + hf * 16 + 8]);
        }
        v8f S0 = {}, S1 = {};
        S0 = wmma_bf16(aq0, kb[0].v, S0);
        S0 = wmma_bf16(aq1, kb[1].v, S0);
        S1 = wmma_bf16(aq0, kb[2].v, S1);
        S1 = wmma_bf16(aq1, kb[3].v, S1);

        // online softmax over this 32-column block
#pragma unroll
        for (int i = 0; i < 8; ++i) {
            float mx = fmaxf(S0[i], S1[i]);
            mx = fmaxf(mx, __shfl_xor(mx, 1, 32));
            mx = fmaxf(mx, __shfl_xor(mx, 2, 32));
            mx = fmaxf(mx, __shfl_xor(mx, 4, 32));
            mx = fmaxf(mx, __shfl_xor(mx, 8, 32));
            float nm = fmaxf(rm[i], mx);
            float sc = __expf(rm[i] - nm);
            float p0 = __expf(S0[i] - nm);
            float p1 = __expf(S1[i] - nm);
            float ps = p0 + p1;
            ps += __shfl_xor(ps, 1, 32);
            ps += __shfl_xor(ps, 2, 32);
            ps += __shfl_xor(ps, 4, 32);
            ps += __shfl_xor(ps, 8, 32);
            rl[i] = rl[i] * sc + ps;
            rm[i] = nm;
#pragma unroll
            for (int t = 0; t < 8; ++t) O[t][i] *= sc;
            // D-layout -> row-major P tile in this wave's private LDS slice
            int row = i + 8 * hf;
            Plds[w][row][r]      = f2bf(p0);
            Plds[w][row][16 + r] = f2bf(p1);
        }

        // reload P as an A fragment (ds_load_b128 x2, in-order with ds stores)
        FragU pf;
        pf.q[0] = *(const uint4*)(&Plds[w][r][hf * 16]);
        pf.q[1] = *(const uint4*)(&Plds[w][r][hf * 16 + 8]);

        // O += P @ V^T over this wave's 128 channels, V from LDS
#pragma unroll
        for (int t = 0; t < 8; ++t) {
            FragU bvf;
            int cl = c0 + t * 16 + r;
            bvf.q[0] = *(const uint4*)(&Vbuf[cur][cl][hf * 16]);
            bvf.q[1] = *(const uint4*)(&Vbuf[cur][cl][hf * 16 + 8]);
            O[t] = wmma_bf16(pf.v, bvf.v, O[t]);
        }
    }

    // epilogue: normalize, gamma * am + x
    const float g = gamma[0];
#pragma unroll
    for (int i = 0; i < 8; ++i) {
        float inv = 1.0f / rl[i];
        int n = n0 + i + 8 * hf;
#pragma unroll
        for (int t = 0; t < 8; ++t) {
            int c = c0 + t * 16 + r;
            size_t idx = ((size_t)b * C_ + c) * N_ + n;
            out[idx] = g * (O[t][i] * inv) + x[idx];
        }
    }
}

// ---------- launcher ----------

extern "C" void kernel_launch(void* const* d_in, const int* in_sizes, int n_in,
                              void* d_out, int out_size, void* d_ws, size_t ws_size,
                              hipStream_t stream) {
    const float* x     = (const float*)d_in[0];
    const float* Wq    = (const float*)d_in[1];
    const float* bq    = (const float*)d_in[2];
    const float* Wk    = (const float*)d_in[3];
    const float* bk    = (const float*)d_in[4];
    const float* Wv    = (const float*)d_in[5];
    const float* bv    = (const float*)d_in[6];
    const float* gamma = (const float*)d_in[7];
    float* out = (float*)d_out;

    unsigned char* ws = (unsigned char*)d_ws;
    size_t off = 0;
    unsigned short* xTb  = (unsigned short*)(ws + off); off += (size_t)B_ * N_ * C_ * 2;
    unsigned short* Wqb  = (unsigned short*)(ws + off); off += (size_t)CQ_ * C_ * 2;
    unsigned short* Wkb  = (unsigned short*)(ws + off); off += (size_t)CQ_ * C_ * 2;
    unsigned short* Wvb  = (unsigned short*)(ws + off); off += (size_t)C_ * C_ * 2;
    unsigned short* qcTb = (unsigned short*)(ws + off); off += (size_t)B_ * N_ * CQ_ * 2;
    unsigned short* kTb  = (unsigned short*)(ws + off); off += (size_t)B_ * N_ * CQ_ * 2;
    unsigned short* vb   = (unsigned short*)(ws + off); off += (size_t)B_ * C_ * N_ * 2;

    cvt_f32_bf16<<<(CQ_ * C_ + 255) / 256, 256, 0, stream>>>(Wq, Wqb, CQ_ * C_);
    cvt_f32_bf16<<<(CQ_ * C_ + 255) / 256, 256, 0, stream>>>(Wk, Wkb, CQ_ * C_);
    cvt_f32_bf16<<<(C_ * C_ + 255) / 256, 256, 0, stream>>>(Wv, Wvb, C_ * C_);

    transpose_x<<<(int)(((size_t)B_ * N_ * C_) / 256), 256, 0, stream>>>(x, xTb);

    qk_proj<<<B_ * (N_ / 16), 32, 0, stream>>>(xTb, Wqb, Wkb, bq, bk, qcTb, kTb);
    v_proj<<<B_ * (C_ / 64) * (N_ / 64), 128, 0, stream>>>(xTb, Wvb, bv, vb);
    attn<<<B_ * (N_ / 16), 128, 0, stream>>>(qcTb, kTb, vb, x, gamma, out);
}
